// Attention_6940667150512
// MI455X (gfx1250) — matrix operand order repaired and checked by reading, not compiled, not run on hardware
//
#include <hip/hip_runtime.h>

// Problem constants
#define BB   512
#define SS   196
#define DD   1024
#define AA   512
#define OO   1000
#define MTOT (BB*SS)   // 100352 = 3136 * 32

typedef __attribute__((ext_vector_type(16))) __bf16 v16bf;
typedef __attribute__((ext_vector_type(8)))  float  v8f;
typedef __attribute__((ext_vector_type(4)))  float  v4f;

__device__ __forceinline__ unsigned short f2bf(float f) {
    unsigned int u = __float_as_uint(f);
    unsigned int r = u + 0x7FFFu + ((u >> 16) & 1u);   // round-to-nearest-even
    return (unsigned short)(r >> 16);
}

__device__ __forceinline__ v8f wmma_bf16(v16bf a, v16bf b, v8f c) {
    return __builtin_amdgcn_wmma_f32_16x16x32_bf16(
        false, a, false, b, (short)0, c, false, false);
}

// ---------------------------------------------------------------------------
// Pack fp32 weight W[K x Nsrc] (row-major) into bf16 WMMA B-fragment layout.
// Fragment (kt, nt): K rows kt*32..+31, N cols nt*16..+15; 32 lanes x 16 bf16
// per fragment (32B/lane contiguous). B layout: half=lane>>4 selects K half,
// nl=lane&15 is N col; vgpr v holds (k = kt*32 + half*16 + 2v, 2v+1).
// ---------------------------------------------------------------------------
__global__ void pack_b_kernel(const float* __restrict__ W, int K, int Nsrc,
                              int Npad, unsigned int* __restrict__ out) {
    int gid  = blockIdx.x * blockDim.x + threadIdx.x;
    int lane = gid & 31;
    int frag = gid >> 5;
    int ntiles = Npad >> 4;
    int totalFrags = (K >> 5) * ntiles;
    if (frag >= totalFrags) return;
    int kt = frag / ntiles;
    int nt = frag - kt * ntiles;
    int half = lane >> 4, nl = lane & 15;
    int n = nt * 16 + nl;
    unsigned int* dst = out + (size_t)frag * 256 + (size_t)lane * 8;
#pragma unroll
    for (int v = 0; v < 8; ++v) {
        int k = kt * 32 + half * 16 + 2 * v;
        float f0 = 0.f, f1 = 0.f;
        if (n < Nsrc) {
            f0 = W[(size_t)k * Nsrc + n];
            f1 = W[(size_t)(k + 1) * Nsrc + n];
        }
        dst[v] = (unsigned int)f2bf(f0) | ((unsigned int)f2bf(f1) << 16);
    }
}

// ---------------------------------------------------------------------------
// Cooperative A-tile stage: 16 rows x 64 K fp32 -> bf16 into LDS in A-fragment
// order (two 16x32 frags). Lane holds row M=lane&15; with h=lane>>4 its 16
// bf16 are K = {16h..16h+15} (contiguous).
// 256 threads; each converts 4 bf16 (one non-temporal 128-bit load).
// ---------------------------------------------------------------------------
__device__ __forceinline__ void stage_A(const float* __restrict__ Am, int M, int K,
                                        int m_base, int k0,
                                        unsigned short* lds_a, int tid) {
    int f    = tid >> 7;
    int rem  = tid & 127;
    int run  = rem >> 1;
    int part = rem & 1;
    int lane = run >> 1;
    int rr   = run & 1;
    int half = lane >> 4;
    int row  = lane & 15;
    int kk   = 16 * half + (rr ? 8 : 0) + part * 4;
    int m    = m_base + row;
    float x0 = 0.f, x1 = 0.f, x2 = 0.f, x3 = 0.f;
    if (m < M) {
        const v4f* p = (const v4f*)(Am + (size_t)m * K + (size_t)(k0 + f * 32 + kk));
        v4f v = __builtin_nontemporal_load(p);
        x0 = v.x; x1 = v.y; x2 = v.z; x3 = v.w;
    }
    unsigned int u0 = (unsigned int)f2bf(x0) | ((unsigned int)f2bf(x1) << 16);
    unsigned int u1 = (unsigned int)f2bf(x2) | ((unsigned int)f2bf(x3) << 16);
    unsigned int* d = (unsigned int*)(lds_a + (f * 512 + lane * 16 + rr * 8 + part * 4));
    d[0] = u0; d[1] = u1;
}

// ---------------------------------------------------------------------------
// bf16-WMMA GEMM with bias: C[M x Nout] = A[M x K] * Bpacked + bias.
// Block = 256 threads = 8 waves; covers 16 rows x 512 cols; wave w covers
// 64 cols (4 N-tiles). grid = (M/16, Npad/512).
// ---------------------------------------------------------------------------
__global__ __launch_bounds__(256)
void gemm_bias_kernel(const float* __restrict__ Amat,
                      const unsigned short* __restrict__ Bp,
                      const float* __restrict__ bias,
                      float* __restrict__ Cout,
                      int M, int K, int Npad, int Nout) {
    __shared__ __align__(32) unsigned short lds_a[1024];
    int tid = threadIdx.x, w = tid >> 5, lane = tid & 31;
    int half = lane >> 4, nl = lane & 15;
    int m_base = blockIdx.x * 16;
    int n0 = blockIdx.y * 512 + w * 64;
    int ntilesN = Npad >> 4;

    // Hoisted B pointer: fragment (kt, nt) lives at (kt*ntilesN + nt)*512 ushorts.
    const size_t kstride = (size_t)ntilesN * 512;           // one kt row of frags
    const unsigned short* bp = Bp + (size_t)(n0 >> 4) * 512 + (size_t)lane * 16;

    v8f acc[4] = {};
    for (int kc = 0; kc < K / 64; ++kc) {
        __syncthreads();
        stage_A(Amat, M, K, m_base, kc * 64, lds_a, tid);
        __syncthreads();
        v16bf a0 = *(const v16bf*)(lds_a + lane * 16);
        v16bf a1 = *(const v16bf*)(lds_a + 512 + lane * 16);
#pragma unroll
        for (int t = 0; t < 4; ++t) {
            v16bf b0 = *(const v16bf*)(bp + (size_t)t * 512);
            v16bf b1 = *(const v16bf*)(bp + kstride + (size_t)t * 512);
            acc[t] = wmma_bf16(a0, b0, acc[t]);
            acc[t] = wmma_bf16(a1, b1, acc[t]);
        }
        bp += 2 * kstride;
    }
#pragma unroll
    for (int t = 0; t < 4; ++t) {
        int col = n0 + t * 16 + nl;
        if (col < Nout) {
            float bv = bias[col];
#pragma unroll
            for (int v = 0; v < 8; ++v) {
                int row = m_base + v + 8 * half;
                if (row < M) Cout[(size_t)row * Nout + col] = acc[t][v] + bv;
            }
        }
    }
}

// ---------------------------------------------------------------------------
// Fused hop-score kernel, 32 rows per block (two m-tiles share every B frag).
// For rows m = b*S+s computes the full A=512-wide i_emb row via WMMA, adds
// q_emb[b,:], tanh, dots with Ws, reduces to one score per row. The i_emb
// tensor ([B,S,A] = 205MB) is never materialized. grid = MTOT/32 blocks.
// ---------------------------------------------------------------------------
__global__ __launch_bounds__(256)
void hop_scores_kernel(const float* __restrict__ img,
                       const unsigned short* __restrict__ Bp,
                       const float* __restrict__ q_emb,
                       const float* __restrict__ Ws,
                       float* __restrict__ scores) {
    __shared__ __align__(32) unsigned short lds_a[2048];  // 2 m-tiles
    __shared__ float lds_part[256];                       // [8 waves][32 rows]
    int tid = threadIdx.x, w = tid >> 5, lane = tid & 31;
    int half = lane >> 4, nl = lane & 15;
    int m_base = blockIdx.x * 32;
    int n0 = w * 64;
    const int ntilesN = AA >> 4;                          // 32
    const size_t kstride = (size_t)ntilesN * 512;
    const unsigned short* bp = Bp + (size_t)(n0 >> 4) * 512 + (size_t)lane * 16;

    v8f acc[2][4] = {};
    for (int kc = 0; kc < DD / 64; ++kc) {
        int k0 = kc * 64;
        __syncthreads();
        stage_A(img, MTOT, DD, m_base,      k0, lds_a,        tid);
        stage_A(img, MTOT, DD, m_base + 16, k0, lds_a + 1024, tid);
        __syncthreads();
        v16bf a00 = *(const v16bf*)(lds_a + lane * 16);
        v16bf a01 = *(const v16bf*)(lds_a + 512 + lane * 16);
        v16bf a10 = *(const v16bf*)(lds_a + 1024 + lane * 16);
        v16bf a11 = *(const v16bf*)(lds_a + 1536 + lane * 16);
#pragma unroll
        for (int t = 0; t < 4; ++t) {
            v16bf b0 = *(const v16bf*)(bp + (size_t)t * 512);
            v16bf b1 = *(const v16bf*)(bp + kstride + (size_t)t * 512);
            acc[0][t] = wmma_bf16(a00, b0, acc[0][t]);
            acc[0][t] = wmma_bf16(a01, b1, acc[0][t]);
            acc[1][t] = wmma_bf16(a10, b0, acc[1][t]);
            acc[1][t] = wmma_bf16(a11, b1, acc[1][t]);
        }
        bp += 2 * kstride;
    }

    // Epilogue: tanh(i_emb + q_emb) . Ws, reduced per row.
#pragma unroll
    for (int mt = 0; mt < 2; ++mt) {
        float part[8];
#pragma unroll
        for (int v = 0; v < 8; ++v) part[v] = 0.f;
#pragma unroll
        for (int t = 0; t < 4; ++t) {
            int col = n0 + t * 16 + nl;          // < 512 always
            float wsv = Ws[col];
#pragma unroll
            for (int v = 0; v < 8; ++v) {
                int m = m_base + mt * 16 + v + 8 * half;   // < MTOT (exact tiling)
                int b = m / SS;
                part[v] += tanhf(acc[mt][t][v] + q_emb[(size_t)b * AA + col]) * wsv;
            }
        }
        // Reduce across the 16 lanes of each half (xor masks < 16 keep the
        // two halves of the wave32 independent).
#pragma unroll
        for (int v = 0; v < 8; ++v) {
            part[v] += __shfl_xor(part[v], 1, 32);
            part[v] += __shfl_xor(part[v], 2, 32);
            part[v] += __shfl_xor(part[v], 4, 32);
            part[v] += __shfl_xor(part[v], 8, 32);
        }
        if (nl == 0) {
#pragma unroll
            for (int v = 0; v < 8; ++v)
                lds_part[w * 32 + mt * 16 + half * 8 + v] = part[v];
        }
    }
    __syncthreads();
    if (tid < 32) {
        float s = 0.f;
#pragma unroll
        for (int ww = 0; ww < 8; ++ww) s += lds_part[ww * 32 + tid];
        scores[(size_t)m_base + tid] = s;   // softmax bias dropped (shift-invariant)
    }
}

// ---------------------------------------------------------------------------
// Softmax over S=196 + attention pooling + residual. One block per batch b.
// u_out[b,d] = q_in[b,d] + sum_s softmax(scores[b,:])[s] * img[b,s,d]
// ---------------------------------------------------------------------------
__global__ __launch_bounds__(256)
void softmax_pool_kernel(const float* __restrict__ scores,
                         const float* __restrict__ img,
                         const float* __restrict__ q_in,
                         float* __restrict__ u_out) {
    __shared__ float red[256];
    __shared__ float p[SS];
    int b = blockIdx.x, tid = threadIdx.x;
    float sv = (tid < SS) ? scores[(size_t)b * SS + tid] : -3.4e38f;
    red[tid] = sv; __syncthreads();
    for (int st = 128; st > 0; st >>= 1) {
        if (tid < st) red[tid] = fmaxf(red[tid], red[tid + st]);
        __syncthreads();
    }
    float mx = red[0]; __syncthreads();
    float e = (tid < SS) ? __expf(sv - mx) : 0.f;
    red[tid] = e; __syncthreads();
    for (int st = 128; st > 0; st >>= 1) {
        if (tid < st) red[tid] += red[tid + st];
        __syncthreads();
    }
    float inv = 1.f / red[0];
    if (tid < SS) p[tid] = e * inv;
    __syncthreads();

    int d0 = tid * 4;   // 256 threads x 4 = D
    float a0 = 0.f, a1 = 0.f, a2 = 0.f, a3 = 0.f;
    const float* base = img + (size_t)b * SS * DD;
#pragma unroll 4
    for (int s = 0; s < SS; ++s) {
        float wv = p[s];
        v4f x = __builtin_nontemporal_load((const v4f*)(base + (size_t)s * DD + d0));
        a0 += wv * x.x; a1 += wv * x.y; a2 += wv * x.z; a3 += wv * x.w;
    }
    const float* qp = q_in + (size_t)b * DD + d0;
    v4f q = *(const v4f*)qp;
    v4f o; o.x = q.x + a0; o.y = q.y + a1; o.z = q.z + a2; o.w = q.w + a3;
    *(v4f*)(u_out + (size_t)b * DD + d0) = o;
}

// ---------------------------------------------------------------------------
extern "C" void kernel_launch(void* const* d_in, const int* in_sizes, int n_in,
                              void* d_out, int out_size, void* d_ws, size_t ws_size,
                              hipStream_t stream) {
    (void)in_sizes; (void)n_in; (void)out_size; (void)ws_size;
    const float* ques = (const float*)d_in[0];
    const float* img  = (const float*)d_in[1];
    const float* W11  = (const float*)d_in[2];
    const float* b11  = (const float*)d_in[3];
    const float* W12  = (const float*)d_in[4];
    const float* W13  = (const float*)d_in[5];
    /* b13 = d_in[6]  : dropped (softmax shift-invariant) */
    const float* W21  = (const float*)d_in[7];
    const float* b21  = (const float*)d_in[8];
    const float* W22  = (const float*)d_in[9];
    const float* W23  = (const float*)d_in[10];
    /* b23 = d_in[11] : dropped */
    const float* Wfc  = (const float*)d_in[12];
    const float* bfc  = (const float*)d_in[13];

    char* ws = (char*)d_ws;
    const size_t MB = 1024 * 1024;
    unsigned short* pW11 = (unsigned short*)(ws + 0 * MB);  // 1 MB each
    unsigned short* pW12 = (unsigned short*)(ws + 1 * MB);
    unsigned short* pW21 = (unsigned short*)(ws + 2 * MB);
    unsigned short* pW22 = (unsigned short*)(ws + 3 * MB);
    unsigned short* pWfc = (unsigned short*)(ws + 4 * MB);  // 2 MB (Npad=1024)
    float* q_emb  = (float*)(ws + 6 * MB);                  // 1 MB
    float* scores = (float*)(ws + 7 * MB);                  // 512 KB
    float* u1     = (float*)(ws + 7 * MB + 512 * 1024);     // 2 MB
    float* u2     = (float*)(ws + 9 * MB + 512 * 1024);     // 2 MB

    // Pre-pack all weights into WMMA B-fragment bf16 layout.
    {
        int frags = (DD / 32) * (AA / 16);            // 1024 frags
        int blocks = (frags * 32 + 255) / 256;
        pack_b_kernel<<<blocks, 256, 0, stream>>>(W11, DD, AA, AA, (unsigned int*)pW11);
        pack_b_kernel<<<blocks, 256, 0, stream>>>(W12, DD, AA, AA, (unsigned int*)pW12);
        pack_b_kernel<<<blocks, 256, 0, stream>>>(W21, DD, AA, AA, (unsigned int*)pW21);
        pack_b_kernel<<<blocks, 256, 0, stream>>>(W22, DD, AA, AA, (unsigned int*)pW22);
        int fragsF = (DD / 32) * (1024 / 16);         // pad O=1000 -> 1024 cols
        int blocksF = (fragsF * 32 + 255) / 256;
        pack_b_kernel<<<blocksF, 256, 0, stream>>>(Wfc, DD, OO, 1024, (unsigned int*)pWfc);
    }

    // Hop 1
    gemm_bias_kernel<<<dim3(BB / 16, 1), 256, 0, stream>>>(ques, pW11, b11, q_emb, BB, DD, AA, AA);
    hop_scores_kernel<<<dim3(MTOT / 32), 256, 0, stream>>>(img, pW12, q_emb, W13, scores);
    softmax_pool_kernel<<<dim3(BB), 256, 0, stream>>>(scores, img, ques, u1);

    // Hop 2
    gemm_bias_kernel<<<dim3(BB / 16, 1), 256, 0, stream>>>(u1, pW21, b21, q_emb, BB, DD, AA, AA);
    hop_scores_kernel<<<dim3(MTOT / 32), 256, 0, stream>>>(img, pW22, q_emb, W23, scores);
    softmax_pool_kernel<<<dim3(BB), 256, 0, stream>>>(scores, img, u1, u2);

    // Final FC: u2 @ Wfc + bfc -> d_out [512 x 1000]
    gemm_bias_kernel<<<dim3(BB / 16, 2), 256, 0, stream>>>(u2, pWfc, bfc, (float*)d_out, BB, DD, 1024, OO);
}
